// CrossAttentionBlock_40510131536522
// MI455X (gfx1250) — compile-verified, matrix operand order checked by reading, not run on hardware
//
#include <hip/hip_runtime.h>
#include <hip/hip_bf16.h>

typedef __attribute__((ext_vector_type(16))) __bf16 v16bf;
typedef __attribute__((ext_vector_type(8)))  __bf16 v8bf;
typedef __attribute__((ext_vector_type(8)))  float  v8f;

#define DIMD   768
#define NHEADS 8
#define HDIM   64
#define INNERD 512
#define FFND   3072
#define BATCH  4
#define NSEQ   2048
#define MSEQ   2048
#define ROWS   (BATCH * NSEQ)   // 8192 rows for both x and context paths

// ---------------------------------------------------------------------------
// WMMA helpers (CDNA5 wave32, 16x16x32 bf16 -> f32)
// ---------------------------------------------------------------------------
__device__ __forceinline__ v8f wmma_bf16(v16bf a, v16bf b, v8f c) {
  // (neg_a, A, neg_b, B, c_mod, C, reuse_a, reuse_b)
  return __builtin_amdgcn_wmma_f32_16x16x32_bf16(false, a, false, b, (short)0, c,
                                                 false, false);
}

// Load one 16x32 bf16 fragment from an LDS tile laid out [row][k] with leading
// dim `ld` (elements). Per ISA 7.12.2: lane L holds row L&15; lanes 0-15 own
// K={0..7,16..23}, lanes 16-31 own K={8..15,24..31}. Two ds_load_b128 per frag.
__device__ __forceinline__ v16bf load_frag(const __bf16* base, int ld) {
  int lane = threadIdx.x & 31;
  int r    = lane & 15;
  int kb   = (lane & 16) ? 8 : 0;
  const __bf16* p = base + r * ld + kb;
  v8bf lo = *(const v8bf*)(p);
  v8bf hi = *(const v8bf*)(p + 16);
  v16bf f;
#pragma unroll
  for (int e = 0; e < 8; ++e) { f[e] = lo[e]; f[e + 8] = hi[e]; }
  return f;
}

// Async DMA copy of 16 bytes global -> LDS (GLOBAL_LOAD_ASYNC_TO_LDS_B128,
// tracked by ASYNCcnt). lds_addr is the wave-logical LDS byte address
// (= low 32 bits of the flat shared pointer, per ISA 10.2 aperture rules).
__device__ __forceinline__ void async_copy_b128(unsigned lds_addr,
                                                const void* gaddr) {
  asm volatile("global_load_async_to_lds_b128 %0, %1, off"
               :: "v"(lds_addr), "v"(gaddr)
               : "memory");
}
__device__ __forceinline__ void wait_async0() {
  asm volatile("s_wait_asynccnt 0" ::: "memory");
}

__device__ __forceinline__ float gelu_exact(float v) {
  return 0.5f * v * (1.0f + erff(v * 0.70710678118654752f));
}

// ---------------------------------------------------------------------------
// fp32 -> bf16 conversion (weights)
// ---------------------------------------------------------------------------
__global__ __launch_bounds__(256) void cvt_bf16_kernel(const float* __restrict__ in,
                                                       __bf16* __restrict__ out,
                                                       int n) {
  int i = blockIdx.x * 256 + threadIdx.x;
  if (i < n) out[i] = (__bf16)in[i];
}

// ---------------------------------------------------------------------------
// LayerNorm over 768 cols, one row per block (256 threads, 3 elems/thread),
// bf16 output.
// ---------------------------------------------------------------------------
__global__ __launch_bounds__(256) void ln_bf16_kernel(const float* __restrict__ X,
                                                      const float* __restrict__ g,
                                                      const float* __restrict__ bt,
                                                      __bf16* __restrict__ out) {
  __shared__ float r1[256];
  __shared__ float r2[256];
  int row = blockIdx.x;
  int t   = threadIdx.x;
  const float* xr = X + (size_t)row * DIMD;
  float v0 = xr[t], v1 = xr[t + 256], v2 = xr[t + 512];
  r1[t] = v0 + v1 + v2;
  r2[t] = v0 * v0 + v1 * v1 + v2 * v2;
  __syncthreads();
#pragma unroll
  for (int off = 128; off > 0; off >>= 1) {
    if (t < off) { r1[t] += r1[t + off]; r2[t] += r2[t + off]; }
    __syncthreads();
  }
  float mu  = r1[0] * (1.0f / DIMD);
  float var = r2[0] * (1.0f / DIMD) - mu * mu;
  float inv = rsqrtf(var + 1e-5f);
  __bf16* orow = out + (size_t)row * DIMD;
  orow[t]       = (__bf16)((v0 - mu) * inv * g[t]       + bt[t]);
  orow[t + 256] = (__bf16)((v1 - mu) * inv * g[t + 256] + bt[t + 256]);
  orow[t + 512] = (__bf16)((v2 - mu) * inv * g[t + 512] + bt[t + 512]);
}

// ---------------------------------------------------------------------------
// Tiled WMMA GEMM: C[M,N] = epi(A[M,K] @ B[K,N])
// Block = 256 threads (8 waves). Tile 128x64, K-stage 64, double-buffered LDS.
// A tiles are staged with async global->LDS DMA (ASYNCcnt); B tiles go through
// registers (in-flight transpose to [n][k]). 8 WMMAs per wave per barrier.
// EPI: 0 = plain bf16 out; 1 = f32 out = acc + bias[col] + resid[row,col];
//      2 = bf16 out = gelu(acc + bias[col])
// ---------------------------------------------------------------------------
template <int EPI>
__global__ __launch_bounds__(256) void gemm_bf16_kernel(
    const __bf16* __restrict__ A, const __bf16* __restrict__ Bm,
    int M, int N, int K,
    const float* __restrict__ bias, const float* __restrict__ resid,
    float* __restrict__ Cf, __bf16* __restrict__ Cb) {
  __shared__ __bf16 As[2][128 * 64];   // [row][k]
  __shared__ __bf16 Bt[2][64 * 64];    // [n][k]  (B staged transposed)
  int n0 = blockIdx.x * 64;
  int r0 = blockIdx.y * 128;
  int t  = threadIdx.x;
  int w  = t >> 5;
  int lane = t & 31;

  const v8f vz = {0.f, 0.f, 0.f, 0.f, 0.f, 0.f, 0.f, 0.f};
  v8f acc[4] = {vz, vz, vz, vz};

  int aRow = t >> 1, aKo = (t & 1) * 32;             // A loader mapping
  int bK   = t >> 2, bN16 = (t & 3) * 16;            // B loader mapping

  v8bf bReg[2];

  auto asyncLoadA = [&](int k0, int buf) {
    const __bf16* ap = A + (size_t)(r0 + aRow) * K + k0 + aKo;
    unsigned lp = (unsigned)(uintptr_t)&As[buf][aRow * 64 + aKo];
#pragma unroll
    for (int e = 0; e < 4; ++e) async_copy_b128(lp + 16u * e, ap + 8 * e);
  };
  auto loadB = [&](int k0) {
    const __bf16* bp = Bm + (size_t)(k0 + bK) * N + n0 + bN16;
#pragma unroll
    for (int e = 0; e < 2; ++e) bReg[e] = *(const v8bf*)(bp + 8 * e);
  };
  auto stageB = [&](int buf) {
#pragma unroll
    for (int v = 0; v < 2; ++v)
#pragma unroll
      for (int e = 0; e < 8; ++e)
        Bt[buf][(bN16 + 8 * v + e) * 64 + bK] = bReg[v][e];
  };

  int nT = K >> 6;  // K / 64
  asyncLoadA(0, 0);
  loadB(0);
  stageB(0);
  for (int it = 0; it < nT; ++it) {
    int buf = it & 1;
    if (it + 1 < nT) loadB((it + 1) << 6);
    if (it + 2 < nT) {
      int k2 = (it + 2) << 6;
      __builtin_prefetch(A + (size_t)(r0 + aRow) * K + k2 + aKo, 0, 1);
      __builtin_prefetch(Bm + (size_t)(k2 + bK) * N + n0 + bN16, 0, 1);
    }
    wait_async0();       // this wave's A DMA for stage `it` has landed
    __syncthreads();     // all waves' A DMA + B ds-stores visible
    if (it + 1 < nT) asyncLoadA((it + 1) << 6, buf ^ 1);  // overlaps WMMAs
#pragma unroll
    for (int kk = 0; kk < 64; kk += 32) {
      v16bf af = load_frag(&As[buf][16 * w * 64 + kk], 64);
#pragma unroll
      for (int n = 0; n < 4; ++n) {
        v16bf bf = load_frag(&Bt[buf][16 * n * 64 + kk], 64);
        acc[n] = wmma_bf16(af, bf, acc[n]);
      }
    }
    if (it + 1 < nT) stageB(buf ^ 1);
  }

  // Epilogue. C/D layout: VGPR r <-> row r + 8*half(lane), lane&15 <-> col.
  int colL = lane & 15;
  int rh   = (lane & 16) ? 8 : 0;
#pragma unroll
  for (int n = 0; n < 4; ++n) {
#pragma unroll
    for (int r = 0; r < 8; ++r) {
      int grow = r0 + 16 * w + r + rh;
      int gcol = n0 + 16 * n + colL;
      float v = acc[n][r];
      if (EPI == 1) {
        v += bias[gcol] + resid[(size_t)grow * N + gcol];
        Cf[(size_t)grow * N + gcol] = v;
      } else if (EPI == 2) {
        v = gelu_exact(v + bias[gcol]);
        Cb[(size_t)grow * N + gcol] = (__bf16)v;
      } else {
        Cb[(size_t)grow * N + gcol] = (__bf16)v;
      }
    }
  }
}

// ---------------------------------------------------------------------------
// Flash attention, one head-tile per block. Block = 128 threads (4 waves),
// 64 query rows per block. Streams keys/values in 64-wide chunks with online
// softmax; all matmuls via WMMA; K tiles staged via async global->LDS DMA.
// Q/K/V/O are [B, seq, INNERD] with head h at cols h*64..h*64+63.
// Score scale = 1/sqrt(64).
// ---------------------------------------------------------------------------
__global__ __launch_bounds__(128) void attn_kernel(
    const __bf16* __restrict__ Q, const __bf16* __restrict__ Kg,
    const __bf16* __restrict__ Vg, __bf16* __restrict__ O) {
  __shared__ __bf16 Qs[64 * 64];   // [i][d]
  __shared__ __bf16 Ks[64 * 64];   // [j][d]   (== [n][k] for S = Q K^T)
  __shared__ __bf16 Vt[64 * 64];   // [d][m']  (== [n][k] for O = P V)
  __shared__ __bf16 Ps[64 * 64];   // [i][m']
  __shared__ float  Sb[64 * 64];
  __shared__ float  alphaS[64];
  __shared__ float  rsumS[64];

  int i0 = blockIdx.x * 64;
  int h  = blockIdx.y;
  int b  = blockIdx.z;
  int t  = threadIdx.x;
  int w  = t >> 5;
  int lane = t & 31;
  int colL = lane & 15;
  int rh   = (lane & 16) ? 8 : 0;

  int jL = t >> 1, dL = (t & 1) * 32;

  // Load Q tile (64 rows x 64 d); 128 threads x 32 elems.
  {
    const __bf16* src = Q + ((size_t)(b * NSEQ) + i0 + jL) * INNERD + h * HDIM + dL;
#pragma unroll
    for (int e = 0; e < 4; ++e) {
      v8bf qv = *(const v8bf*)(src + 8 * e);
      *(v8bf*)&Qs[jL * 64 + dL + 8 * e] = qv;
    }
  }

  float run_max = -3.0e38f;
  float run_sum = 0.0f;
  const v8f vz = {0.f, 0.f, 0.f, 0.f, 0.f, 0.f, 0.f, 0.f};
  v8f oacc[4] = {vz, vz, vz, vz};

  for (int m0 = 0; m0 < MSEQ; m0 += 64) {
    if (m0 + 64 < MSEQ) {  // prefetch next chunk of K/V through L2
      __builtin_prefetch(
          Kg + ((size_t)(b * MSEQ) + m0 + 64 + jL) * INNERD + h * HDIM + dL, 0, 1);
      __builtin_prefetch(
          Vg + ((size_t)(b * MSEQ) + m0 + 64 + jL) * INNERD + h * HDIM + dL, 0, 1);
    }
    __syncthreads();   // previous chunk's WMMA reads of Ks/Vt/Ps done
    { // K tile row-major via async DMA
      const __bf16* src = Kg + ((size_t)(b * MSEQ) + m0 + jL) * INNERD + h * HDIM + dL;
      unsigned lp = (unsigned)(uintptr_t)&Ks[jL * 64 + dL];
#pragma unroll
      for (int e = 0; e < 4; ++e) async_copy_b128(lp + 16u * e, src + 8 * e);
    }
    { // V tile transposed into Vt[d][m'] (register path: needs transpose)
      const __bf16* src = Vg + ((size_t)(b * MSEQ) + m0 + jL) * INNERD + h * HDIM + dL;
#pragma unroll
      for (int e = 0; e < 4; ++e) {
        v8bf vv = *(const v8bf*)(src + 8 * e);
#pragma unroll
        for (int q = 0; q < 8; ++q) Vt[(dL + 8 * e + q) * 64 + jL] = vv[q];
      }
    }
    wait_async0();
    __syncthreads();

    // S = Q K^T for this wave's 16 rows (two k-steps over d)
    v8f s[4] = {vz, vz, vz, vz};
#pragma unroll
    for (int kk = 0; kk < 64; kk += 32) {
      v16bf af = load_frag(Qs + 16 * w * 64 + kk, 64);
#pragma unroll
      for (int n = 0; n < 4; ++n) {
        v16bf bf = load_frag(Ks + 16 * n * 64 + kk, 64);
        s[n] = wmma_bf16(af, bf, s[n]);
      }
    }
#pragma unroll
    for (int n = 0; n < 4; ++n)
#pragma unroll
      for (int r = 0; r < 8; ++r)
        Sb[(16 * w + r + rh) * 64 + 16 * n + colL] = s[n][r] * 0.125f;
    __syncthreads();

    // Online softmax: thread t (< 64) owns query row t.
    if (t < 64) {
      float* srow = Sb + t * 64;
      float cmax = -3.0e38f;
#pragma unroll 8
      for (int j = 0; j < 64; ++j) cmax = fmaxf(cmax, srow[j]);
      float nm    = fmaxf(run_max, cmax);
      float alpha = expf(run_max - nm);
      float ssum  = 0.0f;
#pragma unroll 8
      for (int j = 0; j < 64; ++j) {
        float p = expf(srow[j] - nm);
        Ps[t * 64 + j] = (__bf16)p;
        ssum += p;
      }
      run_sum = run_sum * alpha + ssum;
      run_max = nm;
      alphaS[t] = alpha;
    }
    __syncthreads();

    // Rescale O accumulators, then O += P V
#pragma unroll
    for (int r = 0; r < 8; ++r) {
      float al = alphaS[16 * w + r + rh];
#pragma unroll
      for (int n = 0; n < 4; ++n) oacc[n][r] *= al;
    }
#pragma unroll
    for (int kk = 0; kk < 64; kk += 32) {
      v16bf af = load_frag(Ps + 16 * w * 64 + kk, 64);
#pragma unroll
      for (int n = 0; n < 4; ++n) {
        v16bf bf = load_frag(Vt + 16 * n * 64 + kk, 64);
        oacc[n] = wmma_bf16(af, bf, oacc[n]);
      }
    }
  }

  if (t < 64) rsumS[t] = run_sum;
  __syncthreads();

#pragma unroll
  for (int r = 0; r < 8; ++r) {
    int row = 16 * w + r + rh;
    float inv = 1.0f / rsumS[row];
#pragma unroll
    for (int n = 0; n < 4; ++n) {
      int d = 16 * n + colL;
      O[((size_t)(b * NSEQ) + i0 + row) * INNERD + h * HDIM + d] =
          (__bf16)(oacc[n][r] * inv);
    }
  }
}

// ---------------------------------------------------------------------------
// Launch
// ---------------------------------------------------------------------------
extern "C" void kernel_launch(void* const* d_in, const int* in_sizes, int n_in,
                              void* d_out, int out_size, void* d_ws, size_t ws_size,
                              hipStream_t stream) {
  (void)in_sizes; (void)n_in; (void)out_size; (void)ws_size;
  const float* x   = (const float*)d_in[0];
  const float* ctx = (const float*)d_in[1];
  const float* Wq  = (const float*)d_in[2];
  const float* Wk  = (const float*)d_in[3];
  const float* Wv  = (const float*)d_in[4];
  const float* Wo  = (const float*)d_in[5];
  const float* bo  = (const float*)d_in[6];
  const float* g1  = (const float*)d_in[7];
  const float* b1  = (const float*)d_in[8];
  const float* g2  = (const float*)d_in[9];
  const float* b2  = (const float*)d_in[10];
  const float* W1  = (const float*)d_in[11];
  const float* bf1 = (const float*)d_in[12];
  const float* W2  = (const float*)d_in[13];
  const float* bf2 = (const float*)d_in[14];
  float* out = (float*)d_out;

  char* wp = (char*)d_ws;
  auto alloc = [&](size_t bytes) -> char* {
    char* p = wp;
    wp += (bytes + 255) & ~(size_t)255;
    return p;
  };
  __bf16* Wq16  = (__bf16*)alloc((size_t)DIMD * INNERD * 2);
  __bf16* Wk16  = (__bf16*)alloc((size_t)DIMD * INNERD * 2);
  __bf16* Wv16  = (__bf16*)alloc((size_t)DIMD * INNERD * 2);
  __bf16* Wo16  = (__bf16*)alloc((size_t)INNERD * DIMD * 2);
  __bf16* W116  = (__bf16*)alloc((size_t)DIMD * FFND * 2);
  __bf16* W216  = (__bf16*)alloc((size_t)FFND * DIMD * 2);
  __bf16* nx16  = (__bf16*)alloc((size_t)ROWS * DIMD * 2);
  __bf16* nc16  = (__bf16*)alloc((size_t)ROWS * DIMD * 2);
  __bf16* Q16   = (__bf16*)alloc((size_t)ROWS * INNERD * 2);
  __bf16* K16   = (__bf16*)alloc((size_t)ROWS * INNERD * 2);
  __bf16* V16   = (__bf16*)alloc((size_t)ROWS * INNERD * 2);
  __bf16* O16   = (__bf16*)alloc((size_t)ROWS * INNERD * 2);
  float*  x1    = (float*)alloc((size_t)ROWS * DIMD * 4);
  __bf16* hln16 = (__bf16*)alloc((size_t)ROWS * DIMD * 2);
  __bf16* H16   = (__bf16*)alloc((size_t)ROWS * FFND * 2);

  auto cvt = [&](const float* src, __bf16* dst, int n) {
    cvt_bf16_kernel<<<(n + 255) / 256, 256, 0, stream>>>(src, dst, n);
  };
  cvt(Wq, Wq16, DIMD * INNERD);
  cvt(Wk, Wk16, DIMD * INNERD);
  cvt(Wv, Wv16, DIMD * INNERD);
  cvt(Wo, Wo16, INNERD * DIMD);
  cvt(W1, W116, DIMD * FFND);
  cvt(W2, W216, FFND * DIMD);

  // norm1 for x and context
  ln_bf16_kernel<<<ROWS, 256, 0, stream>>>(x,   g1, b1, nx16);
  ln_bf16_kernel<<<ROWS, 256, 0, stream>>>(ctx, g1, b1, nc16);

  // Q/K/V projections (plain bf16 epilogue)
  {
    dim3 g(INNERD / 64, ROWS / 128);
    gemm_bf16_kernel<0><<<g, 256, 0, stream>>>(nx16, Wq16, ROWS, INNERD, DIMD,
                                               nullptr, nullptr, nullptr, Q16);
    gemm_bf16_kernel<0><<<g, 256, 0, stream>>>(nc16, Wk16, ROWS, INNERD, DIMD,
                                               nullptr, nullptr, nullptr, K16);
    gemm_bf16_kernel<0><<<g, 256, 0, stream>>>(nc16, Wv16, ROWS, INNERD, DIMD,
                                               nullptr, nullptr, nullptr, V16);
  }

  // Attention
  {
    dim3 g(NSEQ / 64, NHEADS, BATCH);
    attn_kernel<<<g, 128, 0, stream>>>(Q16, K16, V16, O16);
  }

  // Output projection + bias + residual -> x1 (f32)
  {
    dim3 g(DIMD / 64, ROWS / 128);
    gemm_bf16_kernel<1><<<g, 256, 0, stream>>>(O16, Wo16, ROWS, DIMD, INNERD,
                                               bo, x, x1, nullptr);
  }

  // norm2
  ln_bf16_kernel<<<ROWS, 256, 0, stream>>>(x1, g2, b2, hln16);

  // FFN1: gelu(h @ W1 + bf1) -> bf16
  {
    dim3 g(FFND / 64, ROWS / 128);
    gemm_bf16_kernel<2><<<g, 256, 0, stream>>>(hln16, W116, ROWS, FFND, DIMD,
                                               bf1, nullptr, nullptr, H16);
  }

  // FFN2: x1 + H @ W2 + bf2 -> d_out (f32)
  {
    dim3 g(DIMD / 64, ROWS / 128);
    gemm_bf16_kernel<1><<<g, 256, 0, stream>>>(H16, W216, ROWS, DIMD, FFND,
                                               bf2, x1, out, nullptr);
  }
}
